// PATOQwen2_5_VLDecoderLayer_45792941310552
// MI455X (gfx1250) — matrix operand order repaired and checked by reading, not compiled, not run on hardware
//
#include <hip/hip_runtime.h>
#include <math.h>

// ---------------------------------------------------------------------------
// Problem constants (Qwen2.5-VL decoder layer)
// ---------------------------------------------------------------------------
constexpr int NT  = 1024;        // tokens
constexpr int Dm  = 3584;        // hidden dim
constexpr int Hh  = 28;          // query heads
constexpr int KVH = 4;           // kv heads
constexpr int HD  = 128;         // head dim
constexpr int Im  = 18944;       // mlp intermediate
constexpr int GR  = Hh / KVH;    // GQA group = 7

typedef __attribute__((ext_vector_type(16))) __bf16 v16bf;
typedef __attribute__((ext_vector_type(8)))  __bf16 v8bf;
typedef __attribute__((ext_vector_type(8)))  float  v8f;
typedef int v4i __attribute__((__vector_size__(16)));   // matches builtin proto

#if defined(__has_builtin)
#if __has_builtin(__builtin_amdgcn_global_load_async_to_lds_b128)
#define ASYNC_LDS 1
#endif
#endif

#ifdef ASYNC_LDS
typedef __attribute__((address_space(3))) unsigned short* ldsptr_t;
#else
typedef unsigned short* ldsptr_t;
#endif

__device__ __forceinline__ unsigned short f2bf(float f) {
  unsigned int u = __float_as_uint(f);
  unsigned int r = u + 0x7FFFu + ((u >> 16) & 1u);   // round-to-nearest-even
  return (unsigned short)(r >> 16);
}

__device__ __forceinline__ v8bf ld8bf(const unsigned short* p) {
  return *(const v8bf*)p;   // 16B aligned LDS read -> ds_load_b128
}

// 16-byte global -> LDS copy. Async (ASYNCcnt) when available, else via regs.
// LDS pointer is pre-cast to AS(3) outside the hot loop (no per-call
// addrspacecast null-check).
__device__ __forceinline__ void cp16(const unsigned short* __restrict__ g,
                                     ldsptr_t l) {
#ifdef ASYNC_LDS
  __builtin_amdgcn_global_load_async_to_lds_b128(
      (__attribute__((address_space(1))) v4i*)g,
      (__attribute__((address_space(3))) v4i*)l, 0, 0);
#else
  *(uint4*)l = *(const uint4*)g;
#endif
}

__device__ __forceinline__ void wait_async() {
#ifdef ASYNC_LDS
#if __has_builtin(__builtin_amdgcn_s_wait_asynccnt)
  __builtin_amdgcn_s_wait_asynccnt(0);
#else
  asm volatile("s_wait_asynccnt 0x0" ::: "memory");
#endif
#endif
}

// ---------------------------------------------------------------------------
// Batched WMMA GEMM: C[z] = alpha * A[z] @ B[z/bDiv]^T (+ bias)
//  A: bf16 bits, row-major M x K (contiguous, lda == K)
//  B: bf16 bits, row-major Nc x K (contiguous, ldb == K)  -> C = A @ B^T
//  C: fp32 or bf16 (CBF16), leading dim ldc
// Block tile 64(M) x 128(N); 8 waves, each 32x32 via 4 WMMAs per 32-K step.
// Double-buffered LDS fed by async global->LDS b128 copies; K-loop unrolled
// x2 so the buffer parity is static (requires K % 64 == 0, true for all
// shapes here: 3584, 128, 1024, 18944).
// ---------------------------------------------------------------------------
template <bool CBF16>
__global__ __launch_bounds__(256) void gemm_bt_wmma(
    const unsigned short* __restrict__ A, const unsigned short* __restrict__ B,
    void* __restrict__ Cv, const float* __restrict__ bias, int K, int ldc,
    long long sA, long long sB, long long sC, int bDiv, float alpha) {
  __shared__ unsigned short sAt[2][64 * 32];    // [buf][row][k]
  __shared__ unsigned short sBt[2][128 * 32];   // [buf][col][k]

  const int t    = threadIdx.x;
  const int lane = t & 31;
  const int wave = t >> 5;
  const int mt   = wave >> 2;          // 0..1
  const int nt   = wave & 3;           // 0..3
  const int m0   = blockIdx.y * 64;
  const int n0   = blockIdx.x * 128;
  const int z    = blockIdx.z;

  // per-thread staging pointers (computed once; K-loop just offsets by k0)
  const unsigned short* aG =
      A + (size_t)z * sA + (size_t)(m0 + (t >> 2)) * K + (t & 3) * 8;
  const unsigned short* bG =
      B + (size_t)(z / bDiv) * sB + (size_t)(n0 + (t >> 1)) * K + (t & 1) * 16;
  // LDS destinations, addrspace-cast hoisted out of the loop
  ldsptr_t aL[2] = {(ldsptr_t)&sAt[0][t * 8], (ldsptr_t)&sAt[1][t * 8]};
  ldsptr_t bL[2] = {(ldsptr_t)&sBt[0][t * 16], (ldsptr_t)&sBt[1][t * 16]};

  v8f acc[2][2];
#pragma unroll
  for (int mi = 0; mi < 2; ++mi)
#pragma unroll
    for (int ni = 0; ni < 2; ++ni)
#pragma unroll
      for (int r = 0; r < 8; ++r) acc[mi][ni][r] = 0.f;

  const int kbA = (lane >> 4) * 8;   // A frag: lanes 0-15 k{0..7,16..23}; 16-31 k{8..15,24..31}
  const int kbB = (lane >> 4) * 8;   // B frag: lanes 0-15 k{0..7,16..23}; 16-31 k{8..15,24..31}
  const int ra0 = (mt * 32 + (lane & 15)) * 32;
  const int rb0 = (nt * 32 + (lane & 15)) * 32;

  auto stage = [&](int p, int k0) {
    cp16(aG + k0, aL[p]);
    cp16(bG + k0, bL[p]);
    cp16(bG + k0 + 8, bL[p] + 8);
  };

  auto compute = [&](const unsigned short* sAp, const unsigned short* sBp) {
    v16bf afr[2], bfr[2];
#pragma unroll
    for (int mi = 0; mi < 2; ++mi) {
      v8bf lo = ld8bf(&sAp[ra0 + mi * 16 * 32 + kbA]);
      v8bf hi = ld8bf(&sAp[ra0 + mi * 16 * 32 + kbA + 16]);
      afr[mi] = __builtin_shufflevector(lo, hi, 0, 1, 2, 3, 4, 5, 6, 7, 8, 9,
                                        10, 11, 12, 13, 14, 15);
    }
#pragma unroll
    for (int ni = 0; ni < 2; ++ni) {
      v8bf lo = ld8bf(&sBp[rb0 + ni * 16 * 32 + kbB]);
      v8bf hi = ld8bf(&sBp[rb0 + ni * 16 * 32 + kbB + 16]);
      bfr[ni] = __builtin_shufflevector(lo, hi, 0, 1, 2, 3, 4, 5, 6, 7, 8, 9,
                                        10, 11, 12, 13, 14, 15);
    }
#pragma unroll
    for (int mi = 0; mi < 2; ++mi)
#pragma unroll
      for (int ni = 0; ni < 2; ++ni)
        acc[mi][ni] = __builtin_amdgcn_wmma_f32_16x16x32_bf16(
            false, afr[mi], false, bfr[ni], (short)0, acc[mi][ni], false, false);
  };

  // software pipeline, static parity (K % 64 == 0)
  stage(0, 0);
  for (int k0 = 0; k0 < K; k0 += 64) {
    wait_async();
    __syncthreads();                 // tile k0 (buf0) visible to all waves
    stage(1, k0 + 32);               // always valid: k0+32 <= K-32
    compute(sAt[0], sBt[0]);
    wait_async();
    __syncthreads();                 // tile k0+32 (buf1) visible
    if (k0 + 64 < K) stage(0, k0 + 64);
    compute(sAt[1], sBt[1]);
  }

  // epilogue: D layout VGPR r -> row r + (lane>=16)*8, col = lane%16
  float* Cf = (float*)Cv + (size_t)z * sC;
  unsigned short* Ch = (unsigned short*)Cv + (size_t)z * sC;
#pragma unroll
  for (int mi = 0; mi < 2; ++mi)
#pragma unroll
    for (int ni = 0; ni < 2; ++ni) {
      int col  = n0 + nt * 32 + ni * 16 + (lane & 15);
      int rowb = m0 + mt * 32 + mi * 16 + ((lane >> 4) * 8);
      float bvl = bias ? bias[col] : 0.f;
#pragma unroll
      for (int r = 0; r < 8; ++r) {
        float v = acc[mi][ni][r] * alpha + bvl;
        if (CBF16) Ch[(size_t)(rowb + r) * ldc + col] = f2bf(v);
        else       Cf[(size_t)(rowb + r) * ldc + col] = v;
      }
    }
}

// ---------------------------------------------------------------------------
// Tiled transpose + fp32 -> bf16: dst[n][k] = bf16(src[k][n]); src is K x Nc.
// grid (Nc/32, K/32), block 256 (32x8).
// ---------------------------------------------------------------------------
__global__ __launch_bounds__(256) void wt_transpose_kernel(
    const float* __restrict__ src, unsigned short* __restrict__ dst, int K,
    int Nc) {
  __shared__ float tile[32][33];
  int tx = threadIdx.x & 31, ty = threadIdx.x >> 5;
  int kb = blockIdx.y * 32, nb = blockIdx.x * 32;
#pragma unroll
  for (int r = 0; r < 4; ++r)
    tile[ty + r * 8][tx] = src[(size_t)(kb + ty + r * 8) * Nc + (nb + tx)];
  __syncthreads();
#pragma unroll
  for (int r = 0; r < 4; ++r)
    dst[(size_t)(nb + ty + r * 8) * K + (kb + tx)] =
        f2bf(tile[tx][ty + r * 8]);
}

// ---------------------------------------------------------------------------
// RMSNorm (pre-attn): out_bf16 = x * rsqrt(mean(x^2)+eps) * w
// ---------------------------------------------------------------------------
__global__ __launch_bounds__(256) void rmsnorm1_kernel(
    const float* __restrict__ x, const float* __restrict__ w,
    unsigned short* __restrict__ out) {
  int row = blockIdx.x, t = threadIdx.x;
  __shared__ float red[256];
  float xv[14], ls = 0.f;
#pragma unroll
  for (int c = 0; c < 14; ++c) {
    int idx = t + c * 256;
    float v = x[(size_t)row * Dm + idx];
    xv[c] = v;
    ls += v * v;
  }
  red[t] = ls;
  __syncthreads();
  for (int s = 128; s > 0; s >>= 1) {
    if (t < s) red[t] += red[t + s];
    __syncthreads();
  }
  float rs = rsqrtf(red[0] / (float)Dm + 1e-6f);
#pragma unroll
  for (int c = 0; c < 14; ++c) {
    int idx = t + c * 256;
    out[(size_t)row * Dm + idx] = f2bf(xv[c] * rs * w[idx]);
  }
}

// ---------------------------------------------------------------------------
// Residual + RMSNorm (pre-MLP): h = x0 + proj; y_bf = rmsnorm(h, w)
// ---------------------------------------------------------------------------
__global__ __launch_bounds__(256) void res_rms2_kernel(
    const float* __restrict__ x0, const float* __restrict__ proj,
    const float* __restrict__ w, float* __restrict__ hbuf,
    unsigned short* __restrict__ ybf) {
  int row = blockIdx.x, t = threadIdx.x;
  __shared__ float red[256];
  float xv[14], ls = 0.f;
#pragma unroll
  for (int c = 0; c < 14; ++c) {
    size_t idx = (size_t)row * Dm + t + c * 256;
    float v = x0[idx] + proj[idx];
    hbuf[idx] = v;
    xv[c] = v;
    ls += v * v;
  }
  red[t] = ls;
  __syncthreads();
  for (int s = 128; s > 0; s >>= 1) {
    if (t < s) red[t] += red[t + s];
    __syncthreads();
  }
  float rs = rsqrtf(red[0] / (float)Dm + 1e-6f);
#pragma unroll
  for (int c = 0; c < 14; ++c) {
    int idx = t + c * 256;
    ybf[(size_t)row * Dm + idx] = f2bf(xv[c] * rs * w[idx]);
  }
}

// ---------------------------------------------------------------------------
// mRoPE (sections 16,24,24 per half) + bf16 repack:
//   q -> [h][n][d], k -> [h][n][d], v -> TRANSPOSED [h][d][n] (for attn@v^T)
// grid: (NT, Hh + 2*KVH), block: HD threads
// ---------------------------------------------------------------------------
__global__ void rope_kernel(const float* __restrict__ qf,
                            const float* __restrict__ kf,
                            const float* __restrict__ vf,
                            const float* __restrict__ cosb,
                            const float* __restrict__ sinb,
                            unsigned short* __restrict__ qb,
                            unsigned short* __restrict__ kb,
                            unsigned short* __restrict__ vb) {
  int n = blockIdx.x, y = blockIdx.y, d = threadIdx.x;
  int dd = d & 63;
  int ax = dd < 16 ? 0 : (dd < 40 ? 1 : 2);
  size_t ci = (size_t)ax * NT * HD + (size_t)n * HD + d;
  float cf = cosb[ci], sf = sinb[ci];
  if (y < Hh) {
    size_t base = (size_t)n * (Hh * HD) + (size_t)y * HD;
    float v = qf[base + d];
    float o = qf[base + (d < 64 ? d + 64 : d - 64)];
    float rh = d < 64 ? -o : o;
    qb[((size_t)y * NT + n) * HD + d] = f2bf(v * cf + rh * sf);
  } else if (y < Hh + KVH) {
    int h = y - Hh;
    size_t base = (size_t)n * (KVH * HD) + (size_t)h * HD;
    float v = kf[base + d];
    float o = kf[base + (d < 64 ? d + 64 : d - 64)];
    float rh = d < 64 ? -o : o;
    kb[((size_t)h * NT + n) * HD + d] = f2bf(v * cf + rh * sf);
  } else {
    int h = y - Hh - KVH;
    vb[((size_t)h * HD + d) * NT + n] =
        f2bf(vf[(size_t)n * (KVH * HD) + (size_t)h * HD + d]);
  }
}

// ---------------------------------------------------------------------------
// Causal + prune-masked softmax; writes attn probabilities as bf16.
// grid: (NT rows, Hh heads), 256 threads, 4 cols/thread
// ---------------------------------------------------------------------------
__global__ __launch_bounds__(256) void softmax_kernel(
    const float* __restrict__ scores, const int* __restrict__ pm,
    unsigned short* __restrict__ attn) {
  int i = blockIdx.x, h = blockIdx.y, t = threadIdx.x;
  const float* srow = scores + ((size_t)h * NT + i) * NT;
  unsigned short* arow = attn + ((size_t)h * NT + i) * NT;
  __shared__ float red[256];
  float sv[4], lmax = -3.0e38f;
#pragma unroll
  for (int it = 0; it < 4; ++it) {
    int j = t + it * 256;
    float s = srow[j];
    sv[it] = s;
    if (j <= i) lmax = fmaxf(lmax, s);
  }
  red[t] = lmax;
  __syncthreads();
  for (int s = 128; s > 0; s >>= 1) {
    if (t < s) red[t] = fmaxf(red[t], red[t + s]);
    __syncthreads();
  }
  float mx = red[0];
  __syncthreads();
  float ev[4], lsum = 0.f;
#pragma unroll
  for (int it = 0; it < 4; ++it) {
    int j = t + it * 256;
    bool keep = (j <= i) && (pm[j] != 0 || j == i);
    float e = keep ? __expf(sv[it] - mx) : 0.f;
    ev[it] = e;
    lsum += e;
  }
  red[t] = lsum;
  __syncthreads();
  for (int s = 128; s > 0; s >>= 1) {
    if (t < s) red[t] += red[t + s];
    __syncthreads();
  }
  float denom = red[0] + 1e-6f;
  float addc = 1e-6f / (float)NT;
#pragma unroll
  for (int it = 0; it < 4; ++it) {
    int j = t + it * 256;
    arow[j] = f2bf((ev[it] + addc) / denom);
  }
}

// ---------------------------------------------------------------------------
// SwiGLU gate: m = silu(g) * u  (bf16 out)
// ---------------------------------------------------------------------------
__global__ __launch_bounds__(256) void silu_mul_kernel(
    const float* __restrict__ g, const float* __restrict__ u,
    unsigned short* __restrict__ m) {
  size_t idx = (size_t)blockIdx.x * 256 + threadIdx.x;
  float gv = g[idx];
  float s = gv / (1.f + __expf(-gv));
  m[idx] = f2bf(s * u[idx]);
}

// ---------------------------------------------------------------------------
// Final residual: out = h + mlp
// ---------------------------------------------------------------------------
__global__ __launch_bounds__(256) void add_kernel(const float* __restrict__ a,
                                                  const float* __restrict__ b,
                                                  float* __restrict__ out) {
  size_t idx = (size_t)blockIdx.x * 256 + threadIdx.x;
  out[idx] = a[idx] + b[idx];
}

// ---------------------------------------------------------------------------
// Launch
// ---------------------------------------------------------------------------
extern "C" void kernel_launch(void* const* d_in, const int* in_sizes, int n_in,
                              void* d_out, int out_size, void* d_ws,
                              size_t ws_size, hipStream_t stream) {
  (void)in_sizes; (void)n_in; (void)out_size; (void)ws_size;
  const float* x    = (const float*)d_in[0];
  const float* cosb = (const float*)d_in[1];
  const float* sinb = (const float*)d_in[2];
  const int*   pm   = (const int*)d_in[3];
  const float* Wq   = (const float*)d_in[4];
  const float* bq   = (const float*)d_in[5];
  const float* Wk   = (const float*)d_in[6];
  const float* bk   = (const float*)d_in[7];
  const float* Wv   = (const float*)d_in[8];
  const float* bv   = (const float*)d_in[9];
  const float* Wo   = (const float*)d_in[10];
  const float* ln1  = (const float*)d_in[11];
  const float* ln2  = (const float*)d_in[12];
  const float* Wg   = (const float*)d_in[13];
  const float* Wu   = (const float*)d_in[14];
  const float* Wd   = (const float*)d_in[15];
  float* out = (float*)d_out;

  // workspace carve-up (256B aligned)
  char* w = (char*)d_ws;
  size_t off = 0;
  auto alloc = [&](size_t bytes) -> void* {
    void* p = w + off;
    off = (off + bytes + 255) & ~(size_t)255;
    return p;
  };
  // transposed bf16 weights (Nc x K)
  unsigned short* WqT = (unsigned short*)alloc((size_t)Dm * Dm * 2);
  unsigned short* WkT = (unsigned short*)alloc((size_t)(KVH * HD) * Dm * 2);
  unsigned short* WvT = (unsigned short*)alloc((size_t)(KVH * HD) * Dm * 2);
  unsigned short* WoT = (unsigned short*)alloc((size_t)Dm * Dm * 2);
  unsigned short* WgT = (unsigned short*)alloc((size_t)Im * Dm * 2);
  unsigned short* WuT = (unsigned short*)alloc((size_t)Im * Dm * 2);
  unsigned short* WdT = (unsigned short*)alloc((size_t)Dm * Im * 2);
  // activations
  unsigned short* xn_bf = (unsigned short*)alloc((size_t)NT * Dm * 2);
  float* q_f  = (float*)alloc((size_t)NT * Hh * HD * 4);
  float* k_f  = (float*)alloc((size_t)NT * KVH * HD * 4);
  float* v_f  = (float*)alloc((size_t)NT * KVH * HD * 4);
  unsigned short* q_bf = (unsigned short*)alloc((size_t)Hh * NT * HD * 2);
  unsigned short* k_bf = (unsigned short*)alloc((size_t)KVH * NT * HD * 2);
  unsigned short* vT_bf = (unsigned short*)alloc((size_t)KVH * HD * NT * 2);
  unsigned short* o_bf = (unsigned short*)alloc((size_t)NT * Dm * 2);
  float* proj = (float*)alloc((size_t)NT * Dm * 4);
  float* hbuf = (float*)alloc((size_t)NT * Dm * 4);
  unsigned short* y_bf = (unsigned short*)alloc((size_t)NT * Dm * 2);
  float* mlpo = (float*)alloc((size_t)NT * Dm * 4);
  float* scores = (float*)alloc((size_t)Hh * NT * NT * 4);
  unsigned short* attn = (unsigned short*)alloc((size_t)Hh * NT * NT * 2);
  float* gbuf = (float*)alloc((size_t)NT * Im * 4);
  float* ubuf = (float*)alloc((size_t)NT * Im * 4);
  unsigned short* m_bf = (unsigned short*)alloc((size_t)NT * Im * 2);

  const float inv_sqrt_hd = 0.08838834764831845f;  // 1/sqrt(128)

  // 0) one-shot weight transpose+quantize to bf16 [Nc x K]
  wt_transpose_kernel<<<dim3(Dm / 32, Dm / 32), 256, 0, stream>>>(Wq, WqT, Dm, Dm);
  wt_transpose_kernel<<<dim3((KVH * HD) / 32, Dm / 32), 256, 0, stream>>>(Wk, WkT, Dm, KVH * HD);
  wt_transpose_kernel<<<dim3((KVH * HD) / 32, Dm / 32), 256, 0, stream>>>(Wv, WvT, Dm, KVH * HD);
  wt_transpose_kernel<<<dim3(Dm / 32, Dm / 32), 256, 0, stream>>>(Wo, WoT, Dm, Dm);
  wt_transpose_kernel<<<dim3(Im / 32, Dm / 32), 256, 0, stream>>>(Wg, WgT, Dm, Im);
  wt_transpose_kernel<<<dim3(Im / 32, Dm / 32), 256, 0, stream>>>(Wu, WuT, Dm, Im);
  wt_transpose_kernel<<<dim3(Dm / 32, Im / 32), 256, 0, stream>>>(Wd, WdT, Im, Dm);

  // 1) RMSNorm -> bf16 activations
  rmsnorm1_kernel<<<NT, 256, 0, stream>>>(x, ln1, xn_bf);

  // 2) QKV projections
  gemm_bt_wmma<false><<<dim3(Dm / 128, NT / 64, 1), 256, 0, stream>>>(
      xn_bf, WqT, q_f, bq, Dm, Dm, 0, 0, 0, 1, 1.f);
  gemm_bt_wmma<false><<<dim3((KVH * HD) / 128, NT / 64, 1), 256, 0, stream>>>(
      xn_bf, WkT, k_f, bk, Dm, KVH * HD, 0, 0, 0, 1, 1.f);
  gemm_bt_wmma<false><<<dim3((KVH * HD) / 128, NT / 64, 1), 256, 0, stream>>>(
      xn_bf, WvT, v_f, bv, Dm, KVH * HD, 0, 0, 0, 1, 1.f);

  // 3) mRoPE + bf16 repack (v transposed per-head)
  rope_kernel<<<dim3(NT, Hh + 2 * KVH), HD, 0, stream>>>(q_f, k_f, v_f, cosb,
                                                         sinb, q_bf, k_bf, vT_bf);

  // 4) scores[h] = (q_h @ k_{h/7}^T) / sqrt(HD)
  gemm_bt_wmma<false><<<dim3(NT / 128, NT / 64, Hh), 256, 0, stream>>>(
      q_bf, k_bf, scores, nullptr, HD, NT, (long long)NT * HD,
      (long long)NT * HD, (long long)NT * NT, GR, inv_sqrt_hd);

  // 5) causal + prune-masked softmax -> bf16 probabilities
  softmax_kernel<<<dim3(NT, Hh), 256, 0, stream>>>(scores, pm, attn);

  // 6) o[:, h*HD:(h+1)*HD] = attn_h @ vT_{h/7}^T  (bf16 out)
  gemm_bt_wmma<true><<<dim3(HD / 128, NT / 64, Hh), 256, 0, stream>>>(
      attn, vT_bf, o_bf, nullptr, NT, Hh * HD, (long long)NT * NT,
      (long long)HD * NT, (long long)HD, GR, 1.f);

  // 7) attention out projection
  gemm_bt_wmma<false><<<dim3(Dm / 128, NT / 64, 1), 256, 0, stream>>>(
      o_bf, WoT, proj, nullptr, Dm, Dm, 0, 0, 0, 1, 1.f);

  // 8) h = x + proj ; y = rmsnorm(h, ln2) -> bf16
  res_rms2_kernel<<<NT, 256, 0, stream>>>(x, proj, ln2, hbuf, y_bf);

  // 9) MLP up projections
  gemm_bt_wmma<false><<<dim3(Im / 128, NT / 64, 1), 256, 0, stream>>>(
      y_bf, WgT, gbuf, nullptr, Dm, Im, 0, 0, 0, 1, 1.f);
  gemm_bt_wmma<false><<<dim3(Im / 128, NT / 64, 1), 256, 0, stream>>>(
      y_bf, WuT, ubuf, nullptr, Dm, Im, 0, 0, 0, 1, 1.f);

  // 10) m = silu(g)*u -> bf16
  silu_mul_kernel<<<((size_t)NT * Im) / 256, 256, 0, stream>>>(gbuf, ubuf, m_bf);

  // 11) down projection
  gemm_bt_wmma<false><<<dim3(Dm / 128, NT / 64, 1), 256, 0, stream>>>(
      m_bf, WdT, mlpo, nullptr, Im, Dm, 0, 0, 0, 1, 1.f);

  // 12) out = h + mlp
  add_kernel<<<((size_t)NT * Dm) / 256, 256, 0, stream>>>(hbuf, mlpo, out);
}
